// TemporalGraphEncoder_82033875353612
// MI455X (gfx1250) — compile-verified, matrix operand order checked by reading, not run on hardware
//
#include <hip/hip_runtime.h>
#include <math.h>

#define NN   5000
#define EE   500000
#define DINc 64
#define DDc  128
#define HHc  8
#define CCc  16
#define LLc  6
#define DFFc 512
#define ETOT (EE + NN)

typedef __attribute__((ext_vector_type(16))) _Float16 v16h;
typedef __attribute__((ext_vector_type(8)))  float    v8f;
typedef __attribute__((ext_vector_type(4)))  float    v4f;

// ---------------------------------------------------------------------------
// WMMA GEMM: C = act(A @ B [+ bias])
//   A: [M,K] f32 row-major
//   B: TRANS_B ? [Ncols,K] : [K,Ncols] f32 row-major
// Block: 256 threads (8 wave32). Output tile 128x64; each wave: 16 rows x 64 cols.
// K staged in 32-chunks through LDS, converted f32->f16, accumulated in f32 via
// v_wmma_f32_16x16x32_f16. Requires Ncols % 64 == 0 and K % 32 == 0 (true here).
// ---------------------------------------------------------------------------
template<int TRANS_B, int RELU, int HAS_BIAS>
__global__ __launch_bounds__(256)
void gemm_wmma(const float* __restrict__ A, const float* __restrict__ B,
               const float* __restrict__ bias, float* __restrict__ Cst,
               int M, int K, int Ncols)
{
    __shared__ _Float16 As[128][40];   // [row][k], padded stride
    __shared__ _Float16 Bs[64][40];    // [col][k] (B transposed in LDS)

    const int tid    = threadIdx.x;
    const int wave   = tid >> 5;
    const int lane   = tid & 31;
    const int rowBlk = blockIdx.x * 128;
    const int colBlk = blockIdx.y * 64;

    v8f acc[4] = {};

    for (int kc = 0; kc < K; kc += 32) {
        // ---- stage A chunk: 128 rows x 32 k (coalesced, 16 floats/thread) ----
        {
            const int r  = tid >> 1;            // 0..127
            const int k0 = (tid & 1) * 16;      // 0 or 16
            int gr = rowBlk + r; if (gr > M - 1) gr = M - 1;
            const float* ap = A + (size_t)gr * K + kc + k0;
            if (kc + 32 < K) __builtin_prefetch(ap + 32, 0, 0);
            #pragma unroll
            for (int i = 0; i < 16; i += 4) {
                v4f v = *reinterpret_cast<const v4f*>(ap + i);
                As[r][k0 + i + 0] = (_Float16)v[0];
                As[r][k0 + i + 1] = (_Float16)v[1];
                As[r][k0 + i + 2] = (_Float16)v[2];
                As[r][k0 + i + 3] = (_Float16)v[3];
            }
        }
        // ---- stage B chunk into Bs[col][k] ----
        if (TRANS_B) {
            const int n  = tid >> 2;            // 0..63
            const int k0 = (tid & 3) * 8;       // 0,8,16,24
            const float* bp = B + (size_t)(colBlk + n) * K + kc + k0;
            #pragma unroll
            for (int i = 0; i < 8; i += 4) {
                v4f v = *reinterpret_cast<const v4f*>(bp + i);
                Bs[n][k0 + i + 0] = (_Float16)v[0];
                Bs[n][k0 + i + 1] = (_Float16)v[1];
                Bs[n][k0 + i + 2] = (_Float16)v[2];
                Bs[n][k0 + i + 3] = (_Float16)v[3];
            }
        } else {
            const int k  = tid >> 3;            // 0..31
            const int n0 = (tid & 7) * 8;       // 0..56
            const float* bp = B + (size_t)(kc + k) * Ncols + colBlk + n0;
            #pragma unroll
            for (int i = 0; i < 8; i += 4) {
                v4f v = *reinterpret_cast<const v4f*>(bp + i);
                Bs[n0 + i + 0][k] = (_Float16)v[0];
                Bs[n0 + i + 1][k] = (_Float16)v[1];
                Bs[n0 + i + 2][k] = (_Float16)v[2];
                Bs[n0 + i + 3][k] = (_Float16)v[3];
            }
        }
        __syncthreads();

        // ---- fragments (CDNA5 16-bit layout: lanes 0-15 K0-7/16-23, 16-31 K8-15/24-31)
        const int mrow  = wave * 16 + (lane & 15);
        const int khalf = (lane >> 4) * 8;
        v16h af;
        {
            const _Float16* ap0 = &As[mrow][khalf];
            #pragma unroll
            for (int i = 0; i < 8; ++i) { af[i] = ap0[i]; af[i + 8] = ap0[16 + i]; }
        }
        #pragma unroll
        for (int t = 0; t < 4; ++t) {
            const _Float16* bp0 = &Bs[t * 16 + (lane & 15)][khalf];
            v16h bf;
            #pragma unroll
            for (int i = 0; i < 8; ++i) { bf[i] = bp0[i]; bf[i + 8] = bp0[16 + i]; }
            acc[t] = __builtin_amdgcn_wmma_f32_16x16x32_f16(
                false, af, false, bf, (short)0, acc[t], false, false);
        }
        __syncthreads();
    }

    // ---- epilogue: C layout vgpr r -> M = r (+8 for upper lane half), N = lane&15
    const int rbase = rowBlk + wave * 16 + ((lane >> 4) << 3);
    const int cIdx  = colBlk + (lane & 15);
    #pragma unroll
    for (int t = 0; t < 4; ++t) {
        const int col = cIdx + t * 16;
        const float bv = HAS_BIAS ? bias[col] : 0.f;
        #pragma unroll
        for (int r = 0; r < 8; ++r) {
            const int row = rbase + r;
            if (row < M) {
                float v = acc[t][r] + bv;
                if (RELU) v = v > 0.f ? v : 0.f;
                Cst[(size_t)row * Ncols + col] = v;
            }
        }
    }
}

// ---------------------------------------------------------------------------
// h[n,:] = LayerNorm(x[n,:] + res[n,:]) * s + b   (one wave32 per row, D=128)
// ---------------------------------------------------------------------------
__global__ __launch_bounds__(256)
void residual_layernorm(const float* __restrict__ x, const float* __restrict__ res,
                        const float* __restrict__ s, const float* __restrict__ b,
                        float* __restrict__ out)
{
    const int wave = threadIdx.x >> 5;
    const int lane = threadIdx.x & 31;
    const int n    = blockIdx.x * 8 + wave;
    if (n >= NN) return;

    float v[4];
    float sum = 0.f;
    #pragma unroll
    for (int i = 0; i < 4; ++i) {
        const int c = lane + 32 * i;
        v[i] = x[(size_t)n * DDc + c] + res[(size_t)n * DDc + c];
        sum += v[i];
    }
    #pragma unroll
    for (int off = 16; off > 0; off >>= 1) sum += __shfl_xor(sum, off, 32);
    const float mean = sum * (1.f / DDc);

    float var = 0.f;
    #pragma unroll
    for (int i = 0; i < 4; ++i) { const float d = v[i] - mean; var += d * d; }
    #pragma unroll
    for (int off = 16; off > 0; off >>= 1) var += __shfl_xor(var, off, 32);
    const float rstd = rsqrtf(var * (1.f / DDc) + 1e-5f);

    #pragma unroll
    for (int i = 0; i < 4; ++i) {
        const int c = lane + 32 * i;
        out[(size_t)n * DDc + c] = (v[i] - mean) * rstd * s[c] + b[c];
    }
}

// ---------------------------------------------------------------------------
// positional encoding add: h[n,d] += (d even ? sin : cos)(n * 10000^-(d&~1)/D)
// ---------------------------------------------------------------------------
__global__ void add_pos(float* __restrict__ h)
{
    const int idx = blockIdx.x * blockDim.x + threadIdx.x;
    if (idx >= NN * DDc) return;
    const int n = idx >> 7, d = idx & 127;
    const float div = __expf((float)(d & ~1) * (-9.21034037198e0f / 128.f));
    const float ang = (float)n * div;
    h[idx] += (d & 1) ? __cosf(ang) : __sinf(ang);
}

// ---------------------------------------------------------------------------
// GAT pieces
// ---------------------------------------------------------------------------
__device__ inline void atomicMaxFloat(float* addr, float val)
{
    if (val >= 0.f) atomicMax((int*)addr, __float_as_int(val));
    else            atomicMin((unsigned int*)addr, (unsigned int)__float_as_int(val));
}

__global__ void gat_scores(const float* __restrict__ glin,
                           const float* __restrict__ asrc,
                           const float* __restrict__ adst,
                           float* __restrict__ as_, float* __restrict__ ad_)
{
    const int idx = blockIdx.x * blockDim.x + threadIdx.x;
    if (idx >= NN * HHc) return;
    const int n = idx >> 3, hd = idx & 7;
    const float* hp = glin + (size_t)n * DDc + hd * CCc;
    const float* wsrc = asrc + hd * CCc;
    const float* wdst = adst + hd * CCc;
    float s1 = 0.f, s2 = 0.f;
    #pragma unroll
    for (int c = 0; c < CCc; ++c) { s1 += hp[c] * wsrc[c]; s2 += hp[c] * wdst[c]; }
    as_[idx] = s1; ad_[idx] = s2;
}

__global__ void gat_init_out(float* __restrict__ out, const float* __restrict__ bias)
{
    const int idx = blockIdx.x * blockDim.x + threadIdx.x;
    if (idx < NN * DDc) out[idx] = bias[idx & 127];
}

__global__ void gat_init_md(float* __restrict__ m, float* __restrict__ den)
{
    const int idx = blockIdx.x * blockDim.x + threadIdx.x;
    if (idx < NN * HHc) { m[idx] = -INFINITY; den[idx] = 0.f; }
}

// pass 1: leaky-relu score, stash into ebuf, atomic max per (dst, head)
__global__ void edge_max(const int* __restrict__ ei,
                         const float* __restrict__ as_, const float* __restrict__ ad_,
                         float* __restrict__ mmax, float* __restrict__ ebuf)
{
    const int e = blockIdx.x * blockDim.x + threadIdx.x;
    if (e >= ETOT) return;
    const int src = (e < EE) ? ei[e] : (e - EE);
    const int dst = (e < EE) ? ei[EE + e] : (e - EE);
    #pragma unroll
    for (int hd = 0; hd < HHc; ++hd) {
        float sc = as_[src * HHc + hd] + ad_[dst * HHc + hd];
        sc = sc > 0.f ? sc : 0.2f * sc;
        ebuf[(size_t)e * HHc + hd] = sc;
        atomicMaxFloat(&mmax[dst * HHc + hd], sc);
    }
}

// pass 2: exp(score - max), atomic denominator sum
__global__ void edge_exp(const int* __restrict__ ei, const float* __restrict__ mmax,
                         float* __restrict__ ebuf, float* __restrict__ den)
{
    const long long idx = (long long)blockIdx.x * blockDim.x + threadIdx.x;
    if (idx >= (long long)ETOT * HHc) return;
    const int e = (int)(idx >> 3), hd = (int)(idx & 7);
    const int dst = (e < EE) ? ei[EE + e] : (e - EE);
    const float ev = __expf(ebuf[idx] - mmax[dst * HHc + hd]);
    ebuf[idx] = ev;
    atomicAdd(&den[dst * HHc + hd], ev);
}

// pass 3: out[dst] += alpha * h[src]
__global__ void edge_aggregate(const int* __restrict__ ei, const float* __restrict__ glin,
                               const float* __restrict__ ebuf, const float* __restrict__ den,
                               float* __restrict__ out)
{
    const long long idx = (long long)blockIdx.x * blockDim.x + threadIdx.x;
    if (idx >= (long long)ETOT * HHc) return;
    const int e = (int)(idx >> 3), hd = (int)(idx & 7);
    const int src = (e < EE) ? ei[e] : (e - EE);
    const int dst = (e < EE) ? ei[EE + e] : (e - EE);
    const float w = ebuf[idx] / (den[dst * HHc + hd] + 1e-16f);
    const float* hp = glin + (size_t)src * DDc + hd * CCc;
    float* op = out + (size_t)dst * DDc + hd * CCc;
    #pragma unroll
    for (int c = 0; c < CCc; ++c) atomicAdd(op + c, hp[c] * w);
}

__global__ void copy_out(const float* __restrict__ h, float* __restrict__ out)
{
    const int idx = blockIdx.x * blockDim.x + threadIdx.x;
    if (idx < NN * DDc) out[idx] = h[idx];
}

// ---------------------------------------------------------------------------
// host side
// ---------------------------------------------------------------------------
static void launch_gemm(int trans_b, int relu, int has_bias,
                        const float* A, const float* B, const float* bias, float* C,
                        int M, int K, int Ncols, hipStream_t s)
{
    dim3 g((M + 127) / 128, Ncols / 64), b(256);
    if (!has_bias) {
        gemm_wmma<0, 0, 0><<<g, b, 0, s>>>(A, B, bias, C, M, K, Ncols);
    } else if (trans_b) {
        gemm_wmma<1, 0, 1><<<g, b, 0, s>>>(A, B, bias, C, M, K, Ncols);
    } else if (relu) {
        gemm_wmma<0, 1, 1><<<g, b, 0, s>>>(A, B, bias, C, M, K, Ncols);
    } else {
        gemm_wmma<0, 0, 1><<<g, b, 0, s>>>(A, B, bias, C, M, K, Ncols);
    }
}

extern "C" void kernel_launch(void* const* d_in, const int* in_sizes, int n_in,
                              void* d_out, int out_size, void* d_ws, size_t ws_size,
                              hipStream_t stream)
{
    (void)in_sizes; (void)n_in; (void)out_size; (void)ws_size;

    const float* x     = (const float*)d_in[0];
    const int*   ei    = (const int*)  d_in[1];
    const float* Wp    = (const float*)d_in[2];
    const float* bp    = (const float*)d_in[3];
    const float* qkv_w = (const float*)d_in[4];
    const float* qkv_b = (const float*)d_in[5];
    const float* out_w = (const float*)d_in[6];
    const float* out_b = (const float*)d_in[7];
    const float* gat_w = (const float*)d_in[8];
    const float* a_src = (const float*)d_in[9];
    const float* a_dst = (const float*)d_in[10];
    const float* gat_b = (const float*)d_in[11];
    const float* w1    = (const float*)d_in[12];
    const float* b1    = (const float*)d_in[13];
    const float* w2    = (const float*)d_in[14];
    const float* b2    = (const float*)d_in[15];
    const float* ln_s  = (const float*)d_in[16];
    const float* ln_b  = (const float*)d_in[17];

    float* ws   = (float*)d_ws;
    float* h    = ws;                          // [N,D]
    float* buf1 = h    + (size_t)NN * DDc;     // [N,D]
    float* buf2 = buf1 + (size_t)NN * DDc;     // [N,D]
    float* mid  = buf2 + (size_t)NN * DDc;     // [N,DFF]
    float* as_  = mid  + (size_t)NN * DFFc;    // [N,H]
    float* ad_  = as_  + (size_t)NN * HHc;     // [N,H]
    float* mmax = ad_  + (size_t)NN * HHc;     // [N,H]
    float* den  = mmax + (size_t)NN * HHc;     // [N,H]
    float* ebuf = den  + (size_t)NN * HHc;     // [Etot,H]

    const int nd_blocks  = (NN * DDc + 255) / 256;
    const int nh_blocks  = (NN * HHc + 255) / 256;
    const int e_blocks   = (ETOT + 255) / 256;
    const int eh_blocks  = (int)(((long long)ETOT * HHc + 255) / 256);
    const int ln_blocks  = (NN + 7) / 8;

    // input projection + positional encoding
    launch_gemm(0, 0, 1, x, Wp, bp, h, NN, DINc, DDc, stream);
    add_pos<<<nd_blocks, 256, 0, stream>>>(h);

    for (int l = 0; l < LLc; ++l) {
        const float* qw  = qkv_w + (size_t)l * 3 * DDc * DDc + 2 * DDc * DDc; // V rows
        const float* qb  = qkv_b + (size_t)l * 3 * DDc + 2 * DDc;
        const float* ow  = out_w + (size_t)l * DDc * DDc;
        const float* ob  = out_b + (size_t)l * DDc;
        const float* gw  = gat_w + (size_t)l * DDc * DDc;
        const float* asw = a_src + (size_t)l * HHc * CCc;
        const float* adw = a_dst + (size_t)l * HHc * CCc;
        const float* gb  = gat_b + (size_t)l * DDc;
        const float* w1l = w1 + (size_t)l * DDc * DFFc;
        const float* b1l = b1 + (size_t)l * DFFc;
        const float* w2l = w2 + (size_t)l * DFFc * DDc;
        const float* b2l = b2 + (size_t)l * DDc;
        const float* ls  = ln_s + (size_t)l * 3 * DDc;
        const float* lb  = ln_b + (size_t)l * 3 * DDc;

        // --- MHSA (seq_len==1): a = (h@Wv.T + bv) @ Wout.T + bo ---
        launch_gemm(1, 0, 1, h,    qw, qb, buf2, NN, DDc, DDc, stream);
        launch_gemm(1, 0, 1, buf2, ow, ob, buf1, NN, DDc, DDc, stream);
        residual_layernorm<<<ln_blocks, 256, 0, stream>>>(h, buf1, ls, lb, h);

        // --- GAT: glin = h @ W (NO bias; bias is added once, post-aggregation) ---
        launch_gemm(0, 0, 0, h, gw, nullptr, buf2, NN, DDc, DDc, stream);
        gat_scores<<<nh_blocks, 256, 0, stream>>>(buf2, asw, adw, as_, ad_);
        gat_init_md<<<nh_blocks, 256, 0, stream>>>(mmax, den);
        edge_max<<<e_blocks, 256, 0, stream>>>(ei, as_, ad_, mmax, ebuf);
        edge_exp<<<eh_blocks, 256, 0, stream>>>(ei, mmax, ebuf, den);
        gat_init_out<<<nd_blocks, 256, 0, stream>>>(buf1, gb);   // seed with gat bias
        edge_aggregate<<<eh_blocks, 256, 0, stream>>>(ei, buf2, ebuf, den, buf1);
        residual_layernorm<<<ln_blocks, 256, 0, stream>>>(h, buf1, ls + DDc, lb + DDc, h);

        // --- FFN ---
        launch_gemm(0, 1, 1, h,   w1l, b1l, mid,  NN, DDc,  DFFc, stream);
        launch_gemm(0, 0, 1, mid, w2l, b2l, buf1, NN, DFFc, DDc,  stream);
        residual_layernorm<<<ln_blocks, 256, 0, stream>>>(h, buf1, ls + 2 * DDc, lb + 2 * DDc, h);
    }

    copy_out<<<nd_blocks, 256, 0, stream>>>(h, (float*)d_out);
}
